// SelfAttention_21595095564390
// MI455X (gfx1250) — matrix operand order repaired and checked by reading, not compiled, not run on hardware
//
#include <hip/hip_runtime.h>

// MI455X / gfx1250, wave32. Flash-attention SAGAN block using
// V_WMMA_F32_16X16X32_F16 throughout.
//
// Layouts (f16 workspace):
//   Qh, Kh : (B, N, 32)  row-major  -> direct global WMMA fragments, no LDS
//   Vh     : (B, 256, N) row-major  -> direct global A-fragments for P·V
//
// Streamed WMMA fragments are double-buffered so the next ds/global load
// clause issues under the current v_wmma instead of serializing on
// s_wait_dscnt 0.

typedef _Float16 half_t;
typedef __attribute__((ext_vector_type(16))) _Float16 v16h;
typedef __attribute__((ext_vector_type(8)))  _Float16 v8h;
typedef __attribute__((ext_vector_type(8)))  float    v8f;

union F16Frag { v16h v; v8h h[2]; };

__device__ __forceinline__ v8f wmma_f16(v16h a, v16h b, v8f c) {
  // 8 args: (neg_a, A, neg_b, B, c_mod, C, reuse_a, reuse_b)
  return __builtin_amdgcn_wmma_f32_16x16x32_f16(false, a, false, b, (short)0, c,
                                                false, false);
}

#define NPIX 4096   // N = H*W
#define CCH  256    // channels
#define DQK  32     // q/k dim

// ---------------------------------------------------------------------------
// Kernel 0: f32 -> f16 conversion for weights
// ---------------------------------------------------------------------------
__global__ void sagan_cvt_f16(const float* __restrict__ src,
                              half_t* __restrict__ dst, int n) {
  int i = blockIdx.x * blockDim.x + threadIdx.x;
  if (i < n) dst[i] = (half_t)src[i];
}

// ---------------------------------------------------------------------------
// Kernel 1: Q/K projection.  Qh/Kh layout: (B, N, 32) f16, bias fused.
// GEMM (32 x 256) @ (256 x 4096) per batch; x tile staged transposed in LDS.
// grid (N/256, B), 256 threads (8 waves).
// wave w: proj = w>>2 (0=Q,1=K), dtile = (w>>1)&1, ngroup = w&1 (8 n-tiles).
// ---------------------------------------------------------------------------
__global__ __launch_bounds__(256) void sagan_proj_qk(
    const float* __restrict__ x,
    const half_t* __restrict__ WqH, const half_t* __restrict__ WkH,
    const float* __restrict__ bq, const float* __restrict__ bk,
    half_t* __restrict__ Qh, half_t* __restrict__ Kh) {
  const int b    = blockIdx.y;
  const int n0   = blockIdx.x * 256;
  const int tid  = threadIdx.x;
  const int lane = tid & 31;
  const int w    = tid >> 5;
  const int m    = lane & 15;   // A-row / B-col / D-col within tile
  const int hg   = lane >> 4;   // half-group

  __shared__ __align__(16) half_t xT[256 * 40];  // [n][c] padded, 20KB

  const int proj  = w >> 2;
  const int dtile = (w >> 1) & 1;
  const int ng    = w & 1;

  const half_t* WH  = proj ? WkH : WqH;
  const float*  bsv = proj ? bk : bq;
  half_t*       op  = proj ? Kh : Qh;

  v8f acc[8];
#pragma unroll
  for (int t = 0; t < 8; ++t) acc[t] = (v8f){};

  const float* xb = x + (size_t)b * CCH * NPIX;

  for (int cb = 0; cb < 8; ++cb) {     // 8 * 32 = 256 K-dim
    __syncthreads();
#pragma unroll
    for (int k = 0; k < 32; ++k) {     // stage 32c x 256n block transposed
      int li = k * 256 + tid;
      int c = li >> 8, n = li & 255;
      xT[n * 40 + c] = (half_t)xb[(size_t)(cb * 32 + c) * NPIX + n0 + n];
    }
    __syncthreads();

    F16Frag a;  // A = W rows (16 x 32), interleaved K layout
    const half_t* pa = WH + (size_t)(dtile * 16 + m) * CCH + cb * 32;
    a.h[0] = *(const v8h*)(pa + 8 * hg);
    a.h[1] = *(const v8h*)(pa + 16 + 8 * hg);

    // B = x^T tiles (32c x 16n), column-contiguous in LDS; double-buffered
    F16Frag bf[2];
    {
      const half_t* pb = &xT[((ng * 8) * 16 + m) * 40 + 8 * hg];
      bf[0].h[0] = *(const v8h*)(pb);
      bf[0].h[1] = *(const v8h*)(pb + 16);
    }
#pragma unroll
    for (int nt = 0; nt < 8; ++nt) {
      if (nt + 1 < 8) {
        const half_t* pb = &xT[((ng * 8 + nt + 1) * 16 + m) * 40 + 8 * hg];
        bf[(nt + 1) & 1].h[0] = *(const v8h*)(pb);
        bf[(nt + 1) & 1].h[1] = *(const v8h*)(pb + 16);
      }
      acc[nt] = wmma_f16(a.v, bf[nt & 1].v, acc[nt]);
    }
  }

  float biasr[8];
#pragma unroll
  for (int r = 0; r < 8; ++r) biasr[r] = bsv[dtile * 16 + r + 8 * hg];

  // (N, 32) layout: 8 consecutive d per lane -> one 16B store per tile
  half_t* ob = op + (size_t)b * NPIX * DQK;
#pragma unroll
  for (int nt = 0; nt < 8; ++nt) {
    int n = n0 + (ng * 8 + nt) * 16 + m;
    v8h pk;
#pragma unroll
    for (int r = 0; r < 8; ++r) pk[r] = (half_t)(acc[nt][r] + biasr[r]);
    *(v8h*)(ob + (size_t)n * DQK + dtile * 16 + 8 * hg) = pk;
  }
}

// ---------------------------------------------------------------------------
// Kernel 2: V projection.  Vh layout: (B, 256, N) f16, bias fused.
// grid (N/64, B), 256 threads; wave w owns c-tiles {2w, 2w+1}, 4 n-tiles.
// ---------------------------------------------------------------------------
__global__ __launch_bounds__(256) void sagan_proj_v(
    const float* __restrict__ x, const half_t* __restrict__ WvH,
    const float* __restrict__ bv, half_t* __restrict__ Vh) {
  const int b    = blockIdx.y;
  const int n0   = blockIdx.x * 64;
  const int tid  = threadIdx.x;
  const int lane = tid & 31;
  const int w    = tid >> 5;
  const int m    = lane & 15;
  const int hg   = lane >> 4;

  __shared__ __align__(16) half_t xT[64 * 40];  // 5KB

  v8f acc[2][4];
#pragma unroll
  for (int ct = 0; ct < 2; ++ct)
#pragma unroll
    for (int nt = 0; nt < 4; ++nt) acc[ct][nt] = (v8f){};

  const float* xb = x + (size_t)b * CCH * NPIX;

  for (int cb = 0; cb < 8; ++cb) {
    __syncthreads();
#pragma unroll
    for (int k = 0; k < 8; ++k) {
      int li = k * 256 + tid;
      int c = li >> 6, n = li & 63;
      xT[n * 40 + c] = (half_t)xb[(size_t)(cb * 32 + c) * NPIX + n0 + n];
    }
    __syncthreads();

    F16Frag a[2];
#pragma unroll
    for (int ct = 0; ct < 2; ++ct) {
      const half_t* pa = WvH + (size_t)((2 * w + ct) * 16 + m) * CCH + cb * 32;
      a[ct].h[0] = *(const v8h*)(pa + 8 * hg);
      a[ct].h[1] = *(const v8h*)(pa + 16 + 8 * hg);
    }
    // double-buffered B tiles
    F16Frag bf[2];
    {
      const half_t* pb = &xT[(m)*40 + 8 * hg];
      bf[0].h[0] = *(const v8h*)(pb);
      bf[0].h[1] = *(const v8h*)(pb + 16);
    }
#pragma unroll
    for (int nt = 0; nt < 4; ++nt) {
      if (nt + 1 < 4) {
        const half_t* pb = &xT[((nt + 1) * 16 + m) * 40 + 8 * hg];
        bf[(nt + 1) & 1].h[0] = *(const v8h*)(pb);
        bf[(nt + 1) & 1].h[1] = *(const v8h*)(pb + 16);
      }
#pragma unroll
      for (int ct = 0; ct < 2; ++ct)
        acc[ct][nt] = wmma_f16(a[ct].v, bf[nt & 1].v, acc[ct][nt]);
    }
  }

  half_t* ob = Vh + (size_t)b * CCH * NPIX;
#pragma unroll
  for (int ct = 0; ct < 2; ++ct) {
    float biasr[8];
#pragma unroll
    for (int r = 0; r < 8; ++r) biasr[r] = bv[(2 * w + ct) * 16 + r + 8 * hg];
#pragma unroll
    for (int nt = 0; nt < 4; ++nt) {
      int n = n0 + nt * 16 + m;
#pragma unroll
      for (int r = 0; r < 8; ++r) {
        int c = (2 * w + ct) * 16 + r + 8 * hg;
        ob[(size_t)c * NPIX + n] = (half_t)(acc[ct][nt][r] + biasr[r]);
      }
    }
  }
}

// ---------------------------------------------------------------------------
// Kernel 3: flash attention + residual.
// grid (N/16, B), 256 threads = 8 waves.  Workgroup owns one 16-row i-tile.
// Computes energy^T (j-rows x i-cols): K slices are A-fragments straight from
// global, Q^T is a B-fragment loaded once; softmax reduction is per-lane +
// one shfl_xor(16).  Per 256-j super-iteration each wave does 2 energy WMMAs
// and 16 P·V WMMAs for its 32 owned channels; only the exp-weight slab lives
// in LDS (2 barriers per iteration).  P·V fragment streams double-buffered.
// ---------------------------------------------------------------------------
__global__ __launch_bounds__(256) void sagan_attn(
    const float* __restrict__ x, const half_t* __restrict__ Qh,
    const half_t* __restrict__ Kh, const half_t* __restrict__ Vh,
    const float* __restrict__ gamma, float* __restrict__ out) {
  const int b    = blockIdx.y;
  const int i0   = blockIdx.x * 16;
  const int tid  = threadIdx.x;
  const int lane = tid & 31;
  const int w    = tid >> 5;
  const int m    = lane & 15;
  const int hg   = lane >> 4;

  __shared__ __align__(16) half_t pT[16 * 264];   // [i][j] padded, 8.25KB
  __shared__ float wmax[8][16];
  __shared__ float wsum[8][16];

  const half_t* Qb = Qh + (size_t)b * NPIX * DQK;   // (N, 32)
  const half_t* Kb = Kh + (size_t)b * NPIX * DQK;   // (N, 32)
  const half_t* Vb = Vh + (size_t)b * CCH * NPIX;   // (C, N)

  // B fragment for energy^T: Q^T (32d x 16i), column i = i0 + m, d contiguous
  F16Frag qb;
  {
    const half_t* qp = Qb + (size_t)(i0 + m) * DQK + 8 * hg;
    qb.h[0] = *(const v8h*)(qp);
    qb.h[1] = *(const v8h*)(qp + 16);
  }

  v8f acc0 = (v8f){}, acc1 = (v8f){};
  float mcur = -3.0e38f, s = 0.f;
  const int cbase = w * 32;
  const half_t* vrow = Vb + (size_t)(cbase + m) * NPIX;

  for (int t = 0; t < 16; ++t) {  // 16 * 256 = 4096 j
    const int j0 = t * 256;

    // hint next super-iteration's K/V lines toward the caches
    if (t + 1 < 16) {
      __builtin_prefetch(Kb + (size_t)(j0 + 256 + w * 32 + m) * DQK, 0, 0);
      __builtin_prefetch(vrow + j0 + 256, 0, 0);
    }

    // ---- produce: energy^T tiles (j-rows x i-cols) for j-slice of wave w --
    v8f e0 = (v8f){}, e1 = (v8f){};
    {
      F16Frag ka0, ka1;  // A = K rows (16j x 32d), straight from global
      const half_t* kp0 = Kb + (size_t)(j0 + w * 32 + m) * DQK;
      ka0.h[0] = *(const v8h*)(kp0 + 8 * hg);
      ka0.h[1] = *(const v8h*)(kp0 + 16 + 8 * hg);
      const half_t* kp1 = kp0 + 16 * DQK;
      ka1.h[0] = *(const v8h*)(kp1 + 8 * hg);
      ka1.h[1] = *(const v8h*)(kp1 + 16 + 8 * hg);
      e0 = wmma_f16(ka0.v, qb.v, e0);
      e1 = wmma_f16(ka1.v, qb.v, e1);
    }
    // reduction axis j is inside the lane: 16 values for column i = m
    float rm = -3.0e38f;
#pragma unroll
    for (int r = 0; r < 8; ++r) rm = fmaxf(rm, fmaxf(e0[r], e1[r]));
    rm = fmaxf(rm, __shfl_xor(rm, 16, 32));
    if (lane < 16) wmax[w][lane] = rm;
    __syncthreads();  // (A) all wave maxes visible; prev p-slab fully read

    float mnew = mcur;
#pragma unroll
    for (int tt = 0; tt < 8; ++tt) mnew = fmaxf(mnew, wmax[tt][m]);
    float alpha = __expf(mcur - mnew);
    mcur = mnew;

    // exp-weights: 8 consecutive j per lane per tile -> packed 16B stores
    float rs = 0.f;
    v8h ph0, ph1;
#pragma unroll
    for (int r = 0; r < 8; ++r) {
      float p0 = __expf(e0[r] - mnew);
      float p1 = __expf(e1[r] - mnew);
      rs += p0 + p1;
      ph0[r] = (half_t)p0;
      ph1[r] = (half_t)p1;
    }
    *(v8h*)&pT[m * 264 + w * 32 + 8 * hg]      = ph0;  // j = w*32+8hg+[0,8)
    *(v8h*)&pT[m * 264 + w * 32 + 16 + 8 * hg] = ph1;  // j = +16
    rs += __shfl_xor(rs, 16, 32);
    if (lane < 16) wsum[w][lane] = rs;
    __syncthreads();  // (B) p-slab + sums visible

    // ---- consume: this wave owns channels [cbase, cbase+32) ----
    float sumnew = 0.f;
#pragma unroll
    for (int tt = 0; tt < 8; ++tt) sumnew += wsum[tt][m];
    s = alpha * s + sumnew;
    acc0 *= alpha;
    acc1 *= alpha;

    // double-buffered fragment streams for the 8-step K-dim (j) loop
    F16Frag pb[2], va0[2], va1[2];
    {
      const half_t* pp = &pT[m * 264 + 8 * hg];
      pb[0].h[0] = *(const v8h*)(pp);
      pb[0].h[1] = *(const v8h*)(pp + 16);
      const half_t* vp0 = vrow + j0;
      va0[0].h[0] = *(const v8h*)(vp0 + 8 * hg);
      va0[0].h[1] = *(const v8h*)(vp0 + 16 + 8 * hg);
      const half_t* vp1 = vp0 + (size_t)16 * NPIX;
      va1[0].h[0] = *(const v8h*)(vp1 + 8 * hg);
      va1[0].h[1] = *(const v8h*)(vp1 + 16 + 8 * hg);
    }
#pragma unroll
    for (int kb = 0; kb < 8; ++kb) {
      const int cur = kb & 1, nxt = cur ^ 1;
      if (kb + 1 < 8) {
        const half_t* pp = &pT[m * 264 + (kb + 1) * 32 + 8 * hg];
        pb[nxt].h[0] = *(const v8h*)(pp);
        pb[nxt].h[1] = *(const v8h*)(pp + 16);
        const half_t* vp0 = vrow + j0 + (kb + 1) * 32;
        va0[nxt].h[0] = *(const v8h*)(vp0 + 8 * hg);
        va0[nxt].h[1] = *(const v8h*)(vp0 + 16 + 8 * hg);
        const half_t* vp1 = vp0 + (size_t)16 * NPIX;
        va1[nxt].h[0] = *(const v8h*)(vp1 + 8 * hg);
        va1[nxt].h[1] = *(const v8h*)(vp1 + 16 + 8 * hg);
      }
      acc0 = wmma_f16(va0[cur].v, pb[cur].v, acc0);
      acc1 = wmma_f16(va1[cur].v, pb[cur].v, acc1);
    }
  }

  // epilogue: normalize, gamma * out + x
  const float g = gamma[0];
  const float rinv = 1.0f / s;  // softmax sum for column i = m
  const float* xb = x + (size_t)b * CCH * NPIX;
  float* ob = out + (size_t)b * CCH * NPIX;
  const int icol = i0 + m;
#pragma unroll
  for (int cti = 0; cti < 2; ++cti) {
    const v8f& a = cti ? acc1 : acc0;
#pragma unroll
    for (int r = 0; r < 8; ++r) {
      int c = cbase + cti * 16 + r + 8 * hg;
      ob[(size_t)c * NPIX + icol] =
          g * a[r] * rinv + xb[(size_t)c * NPIX + icol];
    }
  }
}

// ---------------------------------------------------------------------------
extern "C" void kernel_launch(void* const* d_in, const int* in_sizes, int n_in,
                              void* d_out, int out_size, void* d_ws,
                              size_t ws_size, hipStream_t stream) {
  (void)in_sizes; (void)n_in; (void)out_size; (void)ws_size;
  const float* x     = (const float*)d_in[0];
  const float* Wq    = (const float*)d_in[1];
  const float* bq    = (const float*)d_in[2];
  const float* Wk    = (const float*)d_in[3];
  const float* bk    = (const float*)d_in[4];
  const float* Wv    = (const float*)d_in[5];
  const float* bv    = (const float*)d_in[6];
  const float* gamma = (const float*)d_in[7];
  float* out = (float*)d_out;

  char* ws = (char*)d_ws;
  half_t* Qh  = (half_t*)(ws);                          // 2 MB  (B,N,32)
  half_t* Kh  = (half_t*)(ws + (2ull << 20));           // 2 MB  (B,N,32)
  half_t* Vh  = (half_t*)(ws + (4ull << 20));           // 16 MB (B,256,N)
  half_t* WqH = (half_t*)(ws + (20ull << 20));          // 16 KB
  half_t* WkH = WqH + DQK * CCH;                        // 16 KB
  half_t* WvH = WkH + DQK * CCH;                        // 128 KB

  sagan_cvt_f16<<<32, 256, 0, stream>>>(Wq, WqH, DQK * CCH);
  sagan_cvt_f16<<<32, 256, 0, stream>>>(Wk, WkH, DQK * CCH);
  sagan_cvt_f16<<<256, 256, 0, stream>>>(Wv, WvH, CCH * CCH);

  sagan_proj_qk<<<dim3(NPIX / 256, 8), 256, 0, stream>>>(x, WqH, WkH, bq, bk,
                                                         Qh, Kh);
  sagan_proj_v<<<dim3(NPIX / 64, 8), 256, 0, stream>>>(x, WvH, bv, Vh);
  sagan_attn<<<dim3(NPIX / 16, 8), 256, 0, stream>>>(x, Qh, Kh, Vh, gamma, out);
}
